// GRULayer_32049045963216
// MI455X (gfx1250) — compile-verified, matrix operand order checked by reading, not run on hardware
//
#include <hip/hip_runtime.h>
#include <hip/hip_bf16.h>

// ---------------------------------------------------------------------------
// GRU (Keras reset_after=True) for MI455X / gfx1250, wave32 + WMMA bf16.
//   B=64, T=256, D=1024, U=1024
// Phase 1: xp = x @ kernel + input_bias      (WMMA GEMM, register-pipelined,
//          K-loop fully unrolled to avoid ping-pong register copies)
// Phase 2: 256 sequential fused step kernels, K-split 4 ways per output tile
//          with LDS reduction + fused sigmoid/tanh gates.
// Weights pre-transposed to [3U, K] bf16 so WMMA A and B fragments are two
// contiguous 16-byte loads per lane (gfx1250 16-bit A/B VGPR layout).
// ---------------------------------------------------------------------------

typedef __bf16 bf16_t;
typedef __attribute__((ext_vector_type(16))) __bf16 v16bf;
typedef __attribute__((ext_vector_type(8)))  __bf16 v8bf;
typedef __attribute__((ext_vector_type(8)))  float  v8f;

#define GB   64      // batch
#define GT   256     // timesteps
#define GD   1024    // input dim
#define GU   1024    // units
#define G3U  3072    // 3*U
#define GM   (GB*GT) // 16384 rows of xp

union BF16Frag { v16bf v; struct { v8bf lo; v8bf hi; } p; };

// 16x32 bf16 WMMA fragment row-slice for this lane.
// lanes 0-15: K = k..k+7 and k+16..k+23 ; lanes 16-31: K = k+8..k+15, k+24..k+31
__device__ __forceinline__ v16bf load_frag(const bf16_t* __restrict__ base, int koff) {
    BF16Frag f;
    f.p.lo = *(const v8bf*)(base + koff);
    f.p.hi = *(const v8bf*)(base + koff + 16);
    return f.v;
}

__device__ __forceinline__ v8f wmma_bf16(v16bf a, v16bf b, v8f c) {
    return __builtin_amdgcn_wmma_f32_16x16x32_bf16(false, a, false, b, (short)0, c,
                                                   false, false);
}

// --------------------------- helper kernels --------------------------------

__global__ void f32_to_bf16_kernel(const float* __restrict__ in,
                                   bf16_t* __restrict__ out, int n) {
    int stride = gridDim.x * blockDim.x;
    for (int i = blockIdx.x * blockDim.x + threadIdx.x; i < n; i += stride)
        out[i] = (bf16_t)in[i];
}

// in: row-major [K, N] f32 ; out: row-major [N, K] bf16 (i.e., transposed)
__global__ void transpose_to_bf16_kernel(const float* __restrict__ in,
                                         bf16_t* __restrict__ out,
                                         int K, int N) {
    int total = K * N;
    int stride = gridDim.x * blockDim.x;
    for (int i = blockIdx.x * blockDim.x + threadIdx.x; i < total; i += stride) {
        int nrow = i / K;
        int k    = i - nrow * K;
        out[i] = (bf16_t)in[(size_t)k * N + nrow];
    }
}

__global__ void init_h_kernel(float* __restrict__ hf, bf16_t* __restrict__ hb, int n) {
    int stride = gridDim.x * blockDim.x;
    for (int i = blockIdx.x * blockDim.x + threadIdx.x; i < n; i += stride) {
        hf[i] = 0.0f;
        hb[i] = (bf16_t)0.0f;
    }
}

// --------------------------- phase 1: xp GEMM ------------------------------
// A  : x bf16      [GM, GD] ; Bt : kernel^T [G3U, GD] bf16
// out: xp f32      [GM, G3U]  (+ input_bias)
// Each wave: 16x64 strip, register ping-pong pipelined over K, full unroll.
struct Frags { v16bf a; v16bf b[4]; };

__device__ __forceinline__ void load_set(Frags& f,
                                         const bf16_t* __restrict__ arow,
                                         const bf16_t* __restrict__ brow,
                                         int k, int koff) {
    __builtin_prefetch(arow + k + 128, 0, 1);     // global_prefetch_b8
    f.a = load_frag(arow + k, koff);
#pragma unroll
    for (int c = 0; c < 4; ++c)
        f.b[c] = load_frag(brow + (size_t)(16 * c) * GD + k, koff);
}

__device__ __forceinline__ void mma_set(v8f acc[4], const Frags& f) {
#pragma unroll
    for (int c = 0; c < 4; ++c)
        acc[c] = wmma_bf16(f.a, f.b[c], acc[c]);
}

__global__ __launch_bounds__(256) void xp_gemm_kernel(
    const bf16_t* __restrict__ A,
    const bf16_t* __restrict__ Bt,
    const float*  __restrict__ bias,
    float* __restrict__ out) {
    const int lane = threadIdx.x & 31;
    const int wave = threadIdx.x >> 5;
    const int job  = blockIdx.x * 8 + wave;       // 0 .. 49151
    const int NQ   = G3U / 64;                    // 48 column quads
    const int mt   = job / NQ;
    const int nq   = job - mt * NQ;
    const int m0   = mt * 16;
    const int n0   = nq * 64;

    const int koff = (lane >> 4) << 3;            // 0 or 8
    const bf16_t* arow = A + (size_t)(m0 + (lane & 15)) * GD;
    const bf16_t* brow = Bt + (size_t)(n0 + (lane & 15)) * GD;

    v8f acc[4] = {v8f{}, v8f{}, v8f{}, v8f{}};
    Frags f0, f1;
    // software pipeline: loads for chunk k+32 issued before MMAs for chunk k.
    // Fully unrolled (16 chunk-pairs) so no loop-carried register copies.
    load_set(f0, arow, brow, 0, koff);
#pragma unroll
    for (int it = 0; it < GD / 64 - 1; ++it) {
        const int k = it * 64;
        load_set(f1, arow, brow, k + 32, koff);
        mma_set(acc, f0);
        load_set(f0, arow, brow, k + 64, koff);
        mma_set(acc, f1);
    }
    load_set(f1, arow, brow, GD - 32, koff);
    mma_set(acc, f0);
    mma_set(acc, f1);

    const int hi = lane >> 4;
    const int nc = lane & 15;
#pragma unroll
    for (int c = 0; c < 4; ++c) {
        int n = n0 + 16 * c + nc;
        float bv = bias[n];
#pragma unroll
        for (int i = 0; i < 8; ++i) {
            int m = m0 + i + 8 * hi;               // C/D layout: lanes16-31 -> M+8
            out[(size_t)m * G3U + n] = acc[c][i] + bv;
        }
    }
}

// --------------------------- phase 2: GRU step -----------------------------
// One block (4 waves, 128 threads) per 16x16 output tile; each wave reduces a
// K=256 slice of the three gate GEMMs, partials combined through LDS, gates
// fused in the same kernel. 256 blocks/step -> 8x the parallelism of a
// tile-per-wave scheme and a 4x shorter dependent-load chain per step.
struct SFrags { v16bf a, bz, br, bh; };

__device__ __forceinline__ void load_sset(SFrags& f,
                                          const bf16_t* __restrict__ arow,
                                          const bf16_t* __restrict__ bz,
                                          const bf16_t* __restrict__ br,
                                          const bf16_t* __restrict__ bh,
                                          int k, int koff) {
    f.a  = load_frag(arow + k, koff);
    f.bz = load_frag(bz + k, koff);
    f.br = load_frag(br + k, koff);
    f.bh = load_frag(bh + k, koff);
}

__device__ __forceinline__ void mma_sset(v8f& az, v8f& ar, v8f& ah, const SFrags& f) {
    az = wmma_bf16(f.a, f.bz, az);
    ar = wmma_bf16(f.a, f.br, ar);
    ah = wmma_bf16(f.a, f.bh, ah);
}

__global__ __launch_bounds__(128) void gru_step_kernel(
    const bf16_t* __restrict__ Hb,     // [GB, GU] bf16 h_t
    const float*  __restrict__ Hf,     // [GB, GU] f32  h_t
    const bf16_t* __restrict__ Rt,     // [G3U, GU] rk^T bf16
    const float*  __restrict__ rbias,  // [G3U]
    const float*  __restrict__ xp,     // [GM, G3U]
    float*  __restrict__ Hf_out,
    bf16_t* __restrict__ Hb_out,
    int t) {
    __shared__ float red[3 * 4 * 256];            // [gate][wave][16x16 tile]

    const int lane = threadIdx.x & 31;
    const int wave = threadIdx.x >> 5;            // 0..3 -> K slice
    const int tile = blockIdx.x;                  // 0..255
    const int mt   = tile >> 6;                   // 0..3
    const int nt   = tile & 63;                   // 0..63
    const int m0   = mt * 16;
    const int n0   = nt * 16;
    const int kbeg = wave * 256;

    const int koff = (lane >> 4) << 3;
    const bf16_t* arow = Hb + (size_t)(m0 + (lane & 15)) * GU + kbeg;
    const bf16_t* bz   = Rt + (size_t)(n0 + (lane & 15)) * GU + kbeg;
    const bf16_t* br   = bz + (size_t)GU * GU;
    const bf16_t* bh   = br + (size_t)GU * GU;

    v8f az = {}, ar = {}, ah = {};
    SFrags f0, f1;
    // 8 chunks of K=32, ping-pong pipelined, fully unrolled
    load_sset(f0, arow, bz, br, bh, 0, koff);
#pragma unroll
    for (int it = 0; it < 3; ++it) {              // chunks 0..5
        const int k = it * 64;
        load_sset(f1, arow, bz, br, bh, k + 32, koff);
        mma_sset(az, ar, ah, f0);
        load_sset(f0, arow, bz, br, bh, k + 64, koff);
        mma_sset(az, ar, ah, f1);
    }
    load_sset(f1, arow, bz, br, bh, 224, koff);   // chunk 7
    mma_sset(az, ar, ah, f0);
    mma_sset(az, ar, ah, f1);

    // scatter per-wave partial tiles to LDS: index = (gate*4+wave)*256 + m*16+n
    const int hi = lane >> 4;
    const int nc = lane & 15;
#pragma unroll
    for (int i = 0; i < 8; ++i) {
        int e = (i + 8 * hi) * 16 + nc;
        red[(0 * 4 + wave) * 256 + e] = az[i];
        red[(1 * 4 + wave) * 256 + e] = ar[i];
        red[(2 * 4 + wave) * 256 + e] = ah[i];
    }
    __syncthreads();

    // cross-wave reduction + fused gates: 128 threads x 2 elements
#pragma unroll
    for (int rep = 0; rep < 2; ++rep) {
        int e  = threadIdx.x + rep * 128;         // 0..255
        int b  = m0 + (e >> 4);
        int u  = n0 + (e & 15);
        float rz = red[0 * 1024 + e] + red[0 * 1024 + 256 + e] +
                   red[0 * 1024 + 512 + e] + red[0 * 1024 + 768 + e] + rbias[u];
        float rr = red[1 * 1024 + e] + red[1 * 1024 + 256 + e] +
                   red[1 * 1024 + 512 + e] + red[1 * 1024 + 768 + e] + rbias[GU + u];
        float rh = red[2 * 1024 + e] + red[2 * 1024 + 256 + e] +
                   red[2 * 1024 + 512 + e] + red[2 * 1024 + 768 + e] + rbias[2 * GU + u];
        size_t xrow = ((size_t)b * GT + t) * G3U;
        float z  = 1.0f / (1.0f + __expf(-(xp[xrow + u] + rz)));
        float r  = 1.0f / (1.0f + __expf(-(xp[xrow + GU + u] + rr)));
        float hh = tanhf(xp[xrow + 2 * GU + u] + r * rh);
        float ho = Hf[(size_t)b * GU + u];
        float hn = z * ho + (1.0f - z) * hh;
        Hf_out[(size_t)b * GU + u] = hn;
        Hb_out[(size_t)b * GU + u] = (bf16_t)hn;
    }
}

// ------------------------------- launcher ----------------------------------

extern "C" void kernel_launch(void* const* d_in, const int* in_sizes, int n_in,
                              void* d_out, int out_size, void* d_ws, size_t ws_size,
                              hipStream_t stream) {
    const float* x     = (const float*)d_in[0];   // [B,T,D]
    const float* kern  = (const float*)d_in[1];   // [D,3U]
    const float* rkern = (const float*)d_in[2];   // [U,3U]
    const float* ibias = (const float*)d_in[3];   // [3U]
    const float* rbias = (const float*)d_in[4];   // [3U]
    float* out = (float*)d_out;                   // [B,U]

    // Workspace carve-up (256B aligned). Total ~248 MB.
    char* ws = (char*)d_ws;
    size_t off = 0;
    auto carve = [&](size_t bytes) -> char* {
        char* p = ws + off;
        off = (off + bytes + 255) & ~(size_t)255;
        return p;
    };
    float*  xp    = (float*) carve((size_t)GM * G3U * sizeof(float));
    bf16_t* xbf   = (bf16_t*)carve((size_t)GM * GD * sizeof(bf16_t));
    bf16_t* kt    = (bf16_t*)carve((size_t)G3U * GD * sizeof(bf16_t));
    bf16_t* rt    = (bf16_t*)carve((size_t)G3U * GU * sizeof(bf16_t));
    float*  hf[2] = { (float*) carve((size_t)GB * GU * sizeof(float)),
                      (float*) carve((size_t)GB * GU * sizeof(float)) };
    bf16_t* hb[2] = { (bf16_t*)carve((size_t)GB * GU * sizeof(bf16_t)),
                      (bf16_t*)carve((size_t)GB * GU * sizeof(bf16_t)) };
    (void)ws_size; (void)in_sizes; (void)n_in; (void)out_size;

    // 1) x -> bf16
    f32_to_bf16_kernel<<<8192, 256, 0, stream>>>(x, xbf, GM * GD);
    // 2) transpose-convert weights: [K,3U] f32 -> [3U,K] bf16
    transpose_to_bf16_kernel<<<4096, 256, 0, stream>>>(kern,  kt, GD, G3U);
    transpose_to_bf16_kernel<<<4096, 256, 0, stream>>>(rkern, rt, GU, G3U);
    // 3) input projection GEMM: jobs = (16384/16)*(3072/64) = 49152 waves / 8
    xp_gemm_kernel<<<6144, 256, 0, stream>>>(xbf, kt, ibias, xp);
    // 4) h0 = 0 (both ping-pong buffers, deterministic every call)
    init_h_kernel<<<64, 256, 0, stream>>>(hf[0], hb[0], GB * GU);
    init_h_kernel<<<64, 256, 0, stream>>>(hf[1], hb[1], GB * GU);
    // 5) sequential recurrence: 256 blocks x 128 threads per step
    for (int t = 0; t < GT; ++t) {
        int src = t & 1, dst = src ^ 1;
        float* fout = (t == GT - 1) ? out : hf[dst];
        gru_step_kernel<<<256, 128, 0, stream>>>(hb[src], hf[src], rt, rbias, xp,
                                                 fout, hb[dst], t);
    }
}
